// MambaLayer_3dlocalchunk_36326833389652
// MI455X (gfx1250) — compile-verified, matrix operand order checked by reading, not run on hardware
//
#include <hip/hip_runtime.h>
#include <hip/hip_bf16.h>
#include <cstdint>
#include <cstddef>

// ---------------------------------------------------------------------------
// MI455X (gfx1250) implementation of the 3-branch local-chunk Mamba layer.
// GEMMs run on v_wmma_f32_16x16x32_f16 (wave32, 16x16 tiles, K padded to 32/64),
// templated on (K, N) so fragment loads fold to global_load_b128 and the tile
// loop fully unrolls (no scalar branching in the inner loop).
// The sequential SSM scan runs one workgroup per sequence with h-state in VGPRs.
// Workspace requirement: ~335 MB (fits easily in HBM; hot set fits 192MB L2).
// ---------------------------------------------------------------------------

typedef __attribute__((ext_vector_type(16))) _Float16 v16h;
typedef __attribute__((ext_vector_type(8)))  float    v8f;
typedef __attribute__((ext_vector_type(4)))  unsigned int v4u;

union V16H { v16h h; unsigned int u[8]; v4u q[2]; };

__device__ __forceinline__ float silu_f(float x)     { return x / (1.f + __expf(-x)); }
__device__ __forceinline__ float softplus_f(float x) { return (x > 20.f) ? x : log1pf(__expf(x)); }

// ---------------------------------------------------------------------------
// Convert f32 weight matrix [rows x K] to f16 [rowsPad x Kpad], zero padded.
// ---------------------------------------------------------------------------
__global__ void k_cvtw(const float* __restrict__ src, _Float16* __restrict__ dst,
                       int rows, int K, int rowsPad, int Kpad) {
  int i = blockIdx.x * blockDim.x + threadIdx.x;
  if (i >= rowsPad * Kpad) return;
  int rr = i / Kpad, kk = i - rr * Kpad;
  float v = (rr < rows && kk < K) ? src[rr * K + kk] : 0.f;
  dst[i] = (_Float16)v;
}

// ---------------------------------------------------------------------------
// Build LayerNorm'd token matrix (f16, Kpad=32 features) for one branch.
// Token 0 of each sequence is the LN'd global token gt. stype selects the
// gather permutation (0: pixel-shuffle seq, 1: row-major d=hh, 2: d=ww).
// ---------------------------------------------------------------------------
__global__ void k_build_seq(const float* __restrict__ x, const float* __restrict__ gt,
                            const float* __restrict__ nw, const float* __restrict__ nb,
                            int dmodel, int Lp, int stype, _Float16* __restrict__ xln) {
  size_t g = (size_t)blockIdx.x * blockDim.x + threadIdx.x;
  size_t Ttot = (size_t)64 * Lp;
  if (g >= Ttot) return;
  int n = (int)(g / Lp);
  int t = (int)(g - (size_t)n * Lp);
  int h0 = (n >> 3) * 24, w0 = (n & 7) * 24;

  float v[32];
  if (t == 0) {
    for (int m = 0; m < dmodel; ++m) v[m] = gt[m];
  } else {
    int l = t - 1;
    for (int m = 0; m < dmodel; ++m) {
      int c, hh, ww;
      if (stype == 0) {
        int hwh = l / 48, hww = l - hwh * 48;
        hh = hwh >> 1; ww = hww >> 1;
        c  = m * 4 + (hwh & 1) * 2 + (hww & 1);
      } else if (stype == 1) {
        c = l / 24; ww = l - c * 24; hh = m;
      } else {
        c = l / 24; hh = l - c * 24; ww = m;
      }
      v[m] = x[((size_t)c * 192 + (h0 + hh)) * 192 + (w0 + ww)];
    }
  }
  float mean = 0.f;
  for (int m = 0; m < dmodel; ++m) mean += v[m];
  mean /= (float)dmodel;
  float var = 0.f;
  for (int m = 0; m < dmodel; ++m) { float d = v[m] - mean; var += d * d; }
  var /= (float)dmodel;
  float rs = rsqrtf(var + 1e-5f);
  _Float16* o = xln + g * 32;
  for (int m = 0; m < 32; ++m)
    o[m] = (m < dmodel) ? (_Float16)((v[m] - mean) * rs * nw[m] + nb[m]) : (_Float16)0.f;
}

// ---------------------------------------------------------------------------
// Tall-skinny GEMM: O[T x N] = A[T x K] * W[N x K]^T, f16 in / f32 out.
// Templated on (K, N) -> constant strides, fully unrolled tile loop, fragment
// loads fold into global_load_b128 pairs (K-pairs are dword-contiguous; the
// 4 low-half dwords and 4 high-half dwords of each fragment are consecutive).
// One wave per 16-row tile; A fragment reused across all N tiles.
// K in {32, 64}: 1 or 2 chained v_wmma_f32_16x16x32_f16 per tile.
// T is a multiple of 64; grid.x = T/64, block = 128 (4 waves).
// Fragment packing per CDNA5 ISA 7.12.2.
// ---------------------------------------------------------------------------
template <int K, int N>
__global__ void k_gemm_f16(const _Float16* __restrict__ A, const _Float16* __restrict__ W,
                           float* __restrict__ O, int T) {
  constexpr int K2 = K >> 1;                 // dwords per row
  int lane = threadIdx.x & 31;
  int wv   = threadIdx.x >> 5;
  int m0   = (blockIdx.x * 4 + wv) * 16;
  if (m0 >= T) return;                       // wave-uniform

  int r  = lane & 15;                        // M (A) / N (B) row within tile
  int kb2 = (lane >> 4) << 2;                // dword base: 0 (lanes 0-15) or 4
  int hi = (lane >> 4) << 3;                 // D-row offset for upper half-wave
  const v4u* A4 = (const v4u*)A;
  const v4u* W4 = (const v4u*)W;

  // A fragment(s): per lane, dwords [kb2 .. kb2+3] and [8+kb2 .. 8+kb2+3].
  // With K a constant these are two (four for K=64) b128 loads.
  V16H a0, a1;
  {
    size_t rowq = (size_t)(m0 + r) * (K2 >> 2);   // row offset in v4u units
    a0.q[0] = A4[rowq + (kb2 >> 2)];
    a0.q[1] = A4[rowq + 2 + (kb2 >> 2)];
    if constexpr (K == 64) {
      a1.q[0] = A4[rowq + 4 + (kb2 >> 2)];
      a1.q[1] = A4[rowq + 6 + (kb2 >> 2)];
    }
  }

#pragma unroll
  for (int n0 = 0; n0 < N; n0 += 16) {
    V16H b0, b1;
    size_t rowq = (size_t)(n0 + r) * (K2 >> 2);
    b0.q[0] = W4[rowq + (kb2 >> 2)];
    b0.q[1] = W4[rowq + 2 + (kb2 >> 2)];
    if constexpr (K == 64) {
      b1.q[0] = W4[rowq + 4 + (kb2 >> 2)];
      b1.q[1] = W4[rowq + 6 + (kb2 >> 2)];
    }
    v8f c = {};
    c = __builtin_amdgcn_wmma_f32_16x16x32_f16(false, a0.h, false, b0.h,
                                               (short)0, c, false, false);
    if constexpr (K == 64)
      c = __builtin_amdgcn_wmma_f32_16x16x32_f16(false, a1.h, false, b1.h,
                                                 (short)0, c, false, false);
#pragma unroll
    for (int rr = 0; rr < 8; ++rr)
      O[(size_t)(m0 + rr + hi) * N + (n0 + (lane & 15))] = c[rr];
  }
}

// ---------------------------------------------------------------------------
// Causal depthwise conv (k=4) + SiLU + x_proj (34 outs) + dt_proj + softplus.
// One 64-thread block per token; LDS reduction for the 34 x_proj dots.
// ---------------------------------------------------------------------------
__global__ void k_conv_xproj(const float* __restrict__ inproj, int Ncols, int dinner, int Lp,
                             const float* __restrict__ convw, const float* __restrict__ convb,
                             const float* __restrict__ xprojw,
                             const float* __restrict__ dtw,  const float* __restrict__ dtb,
                             float* __restrict__ xs_act, float* __restrict__ dtbuf,
                             float* __restrict__ bc) {
  size_t g = blockIdx.x;                 // global token
  int c = threadIdx.x;                   // channel (block = 64 threads)
  int t = (int)(g % (size_t)Lp);         // position within its sequence
  __shared__ float sxs[64];
  __shared__ float sdbl[34];

  float acc = 0.f;
  if (c < dinner) {
    acc = convb[c];
#pragma unroll
    for (int k = 0; k < 4; ++k) {
      int tt = t - 3 + k;
      if (tt >= 0) acc += inproj[(g - 3 + k) * (size_t)Ncols + c] * convw[c * 4 + k];
    }
    acc = silu_f(acc);
    sxs[c] = acc;
    xs_act[g * 64 + c] = acc;
  } else {
    sxs[c] = 0.f;
  }
  __syncthreads();
  if (c < 34) {
    float s = 0.f;
    for (int j = 0; j < dinner; ++j) s += sxs[j] * xprojw[c * dinner + j];
    sdbl[c] = s;
    if (c >= 2) bc[g * 32 + (c - 2)] = s;   // Bs in [0,16), Cs in [16,32)
  }
  __syncthreads();
  if (c < dinner) {
    float draw = sdbl[0] * dtw[c * 2 + 0] + sdbl[1] * dtw[c * 2 + 1] + dtb[c];
    dtbuf[g * 64 + c] = softplus_f(draw);
  }
}

// ---------------------------------------------------------------------------
// Sequential selective scan: one workgroup per sequence, one lane per channel.
// h[16] state in VGPRs; B/C staged in LDS each step. Emits the gated f16
// activation (y + u*D) * silu(z) ready for the out_proj WMMA GEMM.
// ---------------------------------------------------------------------------
__global__ void k_scan(const float* __restrict__ inproj, int Ncols, int dinner, int Lp,
                       const float* __restrict__ xs_act, const float* __restrict__ dtbuf,
                       const float* __restrict__ bc,
                       const float* __restrict__ Alog, const float* __restrict__ Dp,
                       _Float16* __restrict__ act) {
  int n = blockIdx.x;
  int c = threadIdx.x;                   // block = 64 threads
  __shared__ float sB[16], sC[16];
  float Ar[16], h[16];
  bool active = (c < dinner);
  if (active) {
#pragma unroll
    for (int s = 0; s < 16; ++s) { Ar[s] = -__expf(Alog[c * 16 + s]); h[s] = 0.f; }
  }
  size_t base = (size_t)n * Lp;
  for (int t = 0; t < Lp; ++t) {
    size_t g = base + t;
    __syncthreads();
    if (c < 32) {
      float v = bc[g * 32 + c];
      if (c < 16) sB[c] = v; else sC[c - 16] = v;
    }
    if (c == 0 && t + 8 < Lp)
      __builtin_prefetch(&bc[(g + 8) * 32], 0, 0);   // global_prefetch_b8
    __syncthreads();
    if (active) {
      float dt = dtbuf[g * 64 + c];
      float u  = xs_act[g * 64 + c];
      float du = dt * u;
      float y  = 0.f;
#pragma unroll
      for (int s = 0; s < 16; ++s) {
        h[s] = h[s] * __expf(dt * Ar[s]) + du * sB[s];
        y += h[s] * sC[s];
      }
      float z = inproj[g * (size_t)Ncols + dinner + c];
      act[g * 64 + c] = (_Float16)((y + u * Dp[c]) * silu_f(z));
    } else {
      act[g * 64 + c] = (_Float16)0.f;   // zero pad channels for K=64 GEMM
    }
  }
}

// ---------------------------------------------------------------------------
// Final combine: gather the three branch outputs through the reference's
// fold index math and average into d_out (f32, shape 1x128x192x192).
// ---------------------------------------------------------------------------
__global__ void k_combine(const float* __restrict__ y0, const float* __restrict__ y1,
                          const float* __restrict__ y2, float* __restrict__ out) {
  size_t idx = (size_t)blockIdx.x * blockDim.x + threadIdx.x;
  if (idx >= (size_t)128 * 192 * 192) return;
  int c    = (int)(idx / 36864);
  int rem  = (int)(idx - (size_t)c * 36864);     // h*192 + w
  int nc   = rem / 576;
  int r2m  = rem - nc * 576;
  int hh   = r2m / 24;
  int ww   = r2m - hh * 24;

  int r1 = (c & 3) >> 1, r2b = c & 1;
  size_t g0 = ((size_t)nc * 2305 + 1 + (size_t)(hh * 2 + r1) * 48 + (ww * 2 + r2b)) * 32 + (c >> 2);
  size_t g1 = ((size_t)nc * 3073 + 1 + (size_t)c * 24 + ww) * 32 + hh;
  size_t g2 = ((size_t)nc * 3073 + 1 + (size_t)c * 24 + hh) * 32 + ww;
  out[idx] = (y0[g0] + y1[g1] + y2[g2]) * (1.f / 3.f);
}

// ---------------------------------------------------------------------------
// Host launcher
// ---------------------------------------------------------------------------
extern "C" void kernel_launch(void* const* d_in, const int* in_sizes, int n_in,
                              void* d_out, int out_size, void* d_ws, size_t ws_size,
                              hipStream_t stream) {
  (void)in_sizes; (void)n_in; (void)out_size; (void)ws_size;

  const float* x       = (const float*)d_in[0];
  const float* norm_w  = (const float*)d_in[1];
  const float* norm_b  = (const float*)d_in[2];
  const float* norm2_w = (const float*)d_in[3];
  const float* norm2_b = (const float*)d_in[4];
  const float* gt1     = (const float*)d_in[5];
  const float* gt2     = (const float*)d_in[6];
  // m1: 7..15 ; m2: 16..24  (in_proj_w, conv_w, conv_b, x_proj_w, dt_proj_w,
  //                          dt_proj_b, A_log, D, out_proj_w)
  const float* m1_inw  = (const float*)d_in[7];
  const float* m1_cw   = (const float*)d_in[8];
  const float* m1_cb   = (const float*)d_in[9];
  const float* m1_xw   = (const float*)d_in[10];
  const float* m1_dtw  = (const float*)d_in[11];
  const float* m1_dtb  = (const float*)d_in[12];
  const float* m1_al   = (const float*)d_in[13];
  const float* m1_D    = (const float*)d_in[14];
  const float* m1_ow   = (const float*)d_in[15];
  const float* m2_inw  = (const float*)d_in[16];
  const float* m2_cw   = (const float*)d_in[17];
  const float* m2_cb   = (const float*)d_in[18];
  const float* m2_xw   = (const float*)d_in[19];
  const float* m2_dtw  = (const float*)d_in[20];
  const float* m2_dtb  = (const float*)d_in[21];
  const float* m2_al   = (const float*)d_in[22];
  const float* m2_D    = (const float*)d_in[23];
  const float* m2_ow   = (const float*)d_in[24];

  const int    Lp0 = 2305, Lp1 = 3073;
  const size_t T0  = 64 * (size_t)Lp0;     // 147520 (mult of 64)
  const size_t T1  = 64 * (size_t)Lp1;     // 196672 (mult of 64)
  const size_t Tmax = T1;

  // Workspace layout (256B aligned regions).
  char* ws = (char*)d_ws;
  size_t off = 0;
  auto take = [&](size_t bytes) { size_t s = off; off = (off + bytes + 255) & ~(size_t)255; return s; };
  _Float16* wIn1  = (_Float16*)(ws + take(128 * 32 * 2));
  _Float16* wIn2  = (_Float16*)(ws + take( 96 * 32 * 2));
  _Float16* wOut1 = (_Float16*)(ws + take( 32 * 64 * 2));
  _Float16* wOut2 = (_Float16*)(ws + take( 32 * 64 * 2));
  _Float16* xln   = (_Float16*)(ws + take(Tmax * 32 * 2));
  float*    inpr  = (float*)   (ws + take(Tmax * 128 * 4));
  float*    xsbuf = (float*)   (ws + take(Tmax * 64 * 4));
  float*    dtbuf = (float*)   (ws + take(Tmax * 64 * 4));
  float*    bcbuf = (float*)   (ws + take(Tmax * 32 * 4));
  _Float16* actb  = (_Float16*)(ws + take(Tmax * 64 * 2));
  float*    y0    = (float*)   (ws + take(T0 * 32 * 4));
  float*    y1    = (float*)   (ws + take(T1 * 32 * 4));
  float*    y2    = (float*)   (ws + take(T1 * 32 * 4));

  // 1. Weight conversion (f32 -> padded f16).
  k_cvtw<<<(128 * 32 + 255) / 256, 256, 0, stream>>>(m1_inw, wIn1, 128, 32, 128, 32);
  k_cvtw<<<( 96 * 32 + 255) / 256, 256, 0, stream>>>(m2_inw, wIn2,  96, 24,  96, 32);
  k_cvtw<<<( 32 * 64 + 255) / 256, 256, 0, stream>>>(m1_ow,  wOut1, 32, 64,  32, 64);
  k_cvtw<<<( 32 * 64 + 255) / 256, 256, 0, stream>>>(m2_ow,  wOut2, 24, 48,  32, 64);

  struct Br {
    int dmodel, dinner, Lp, stype, Ninp; size_t T;
    const float *gt, *nw, *nb, *cw, *cb, *xw, *dtw, *dtb, *al, *Dp;
    _Float16 *win, *wout; float* ybuf;
  };
  Br brs[3] = {
    {32, 64, Lp0, 0, 128, T0, gt1, norm_w,  norm_b,  m1_cw, m1_cb, m1_xw, m1_dtw, m1_dtb, m1_al, m1_D, wIn1, wOut1, y0},
    {24, 48, Lp1, 1,  96, T1, gt2, norm2_w, norm2_b, m2_cw, m2_cb, m2_xw, m2_dtw, m2_dtb, m2_al, m2_D, wIn2, wOut2, y1},
    {24, 48, Lp1, 2,  96, T1, gt2, norm2_w, norm2_b, m2_cw, m2_cb, m2_xw, m2_dtw, m2_dtb, m2_al, m2_D, wIn2, wOut2, y2},
  };

  for (int i = 0; i < 3; ++i) {
    const Br& b = brs[i];
    // 2a. Gather + LayerNorm -> f16 tokens
    k_build_seq<<<(unsigned)((b.T + 255) / 256), 256, 0, stream>>>(
        x, b.gt, b.nw, b.nb, b.dmodel, b.Lp, b.stype, xln);
    // 2b. in_proj GEMM (WMMA, K=32), N = 128 (m1) or 96 (m2)
    if (i == 0)
      k_gemm_f16<32, 128><<<(unsigned)(b.T / 64), 128, 0, stream>>>(xln, b.win, inpr, (int)b.T);
    else
      k_gemm_f16<32,  96><<<(unsigned)(b.T / 64), 128, 0, stream>>>(xln, b.win, inpr, (int)b.T);
    // 2c. conv + silu + x_proj + dt
    k_conv_xproj<<<(unsigned)b.T, 64, 0, stream>>>(
        inpr, b.Ninp, b.dinner, b.Lp, b.cw, b.cb, b.xw, b.dtw, b.dtb,
        xsbuf, dtbuf, bcbuf);
    // 2d. sequential selective scan + gating -> f16 activations
    k_scan<<<64, 64, 0, stream>>>(
        inpr, b.Ninp, b.dinner, b.Lp, xsbuf, dtbuf, bcbuf, b.al, b.Dp, actb);
    // 2e. out_proj GEMM (WMMA, K=64, N=32)
    k_gemm_f16<64, 32><<<(unsigned)(b.T / 64), 128, 0, stream>>>(actb, b.wout, b.ybuf, (int)b.T);
  }

  // 3. Combine all branches into the output tensor.
  const size_t NOUT = (size_t)128 * 192 * 192;
  k_combine<<<(unsigned)((NOUT + 255) / 256), 256, 0, stream>>>(y0, y1, y2, (float*)d_out);
}
